// MKAAttention_23648089932463
// MI455X (gfx1250) — matrix operand order repaired and checked by reading, not compiled, not run on hardware
//
#include <hip/hip_runtime.h>

// ---------------------------------------------------------------- constants
constexpr int Bc  = 2;
constexpr int Sc  = 2048;
constexpr int Hc  = 1024;
constexpr int NHc = 16;
constexpr int HDc = 64;
constexpr int Wc  = 16;      // sliding window
constexpr int RDc = 16;      // rope dims
constexpr float SCALEc   = 0.125f;    // 64^-0.5
constexpr float ROPEBASE = 10000.0f;

// ---------------------------------------------------------------- types
typedef __attribute__((ext_vector_type(16))) __bf16 v16bf;
typedef __attribute__((ext_vector_type(8)))  __bf16 bf16x8;
typedef __attribute__((ext_vector_type(8)))  float  v8f;

__device__ __forceinline__ __bf16 f2bf(float f) {
  unsigned u = __float_as_uint(f);
  unsigned r = (u + 0x7FFFu + ((u >> 16) & 1u)) >> 16;
  unsigned short s = (unsigned short)r;
  return __builtin_bit_cast(__bf16, s);
}

__device__ __forceinline__ v8f wmma_bf16(v16bf a, v16bf b, v8f c) {
  // v_wmma_f32_16x16x32_bf16: D = A(16x32) * B(32x16) + C(16x16 f32)
  return __builtin_amdgcn_wmma_f32_16x16x32_bf16(
      /*neg_a=*/false, a, /*neg_b=*/false, b,
      /*c_mod=*/(short)0, c, /*reuse_a=*/false, /*reuse_b=*/false);
}

// A-matrix fragment (16x32 bf16): lane m = lane&15, half = lane>>4.
// elems 0..7  = K [k0 + half*8,      +8)
// elems 8..15 = K [k0 + 16 + half*8, +8)
__device__ __forceinline__ v16bf load_a_frag(const __bf16* base, size_t stride,
                                             int lane, int k0) {
  const int m = lane & 15, h = lane >> 4;
  const __bf16* p = base + (size_t)m * stride + k0 + h * 8;
  union { v16bf v; bf16x8 q[2]; } u;
  u.q[0] = *reinterpret_cast<const bf16x8*>(p);
  u.q[1] = *reinterpret_cast<const bf16x8*>(p + 16);
  return u.v;
}

// B-matrix fragment (32x16 bf16): lane n = lane&15 holds column n.
// elems 0..7 = K [k0 + half*8, +8), elems 8..15 = K [k0 + 16 + half*8, +8)
__device__ __forceinline__ v16bf load_b_frag(const __bf16* base, size_t stride,
                                             int lane, int k0) {
  const int n = lane & 15, h = lane >> 4;
  const __bf16* p = base + (size_t)n * stride + k0 + h * 8;
  union { v16bf v; bf16x8 q[2]; } u;
  u.q[0] = *reinterpret_cast<const bf16x8*>(p);
  u.q[1] = *reinterpret_cast<const bf16x8*>(p + 16);
  return u.v;
}

// ---------------------------------------------------------------- fp32 -> bf16
__global__ void cvt_bf16_kernel(const float* __restrict__ in,
                                __bf16* __restrict__ out, int n) {
  int i = blockIdx.x * blockDim.x + threadIdx.x;
  if (i < n) out[i] = f2bf(in[i]);
}

// ---------------------------------------------------------------- bf16 WMMA GEMM
// C(M,N) = A(M,K) @ W(N,K)^T + bias(N)
// block = 256 threads = 8 waves; block tile 128x128; K-step 32.
// Double-buffered LDS staging of A/B tiles (one s_barrier per K-step).
// MODE 0: fp32 row-major out (ldc = Ndim)
// MODE 1: bf16 head-major out (b, h, s, d), assumes M = Bc*Sc, N = Hc
template <int MODE>
__global__ __launch_bounds__(256)
void gemm_bf16_kernel(const __bf16* __restrict__ A, const __bf16* __restrict__ Wt,
                      const float* __restrict__ bias, void* __restrict__ out,
                      int Mdim, int Ndim, int Kdim) {
  constexpr int LSTR = 40;  // padded LDS row stride (elems); 80B = 20 dword banks
  __shared__ __align__(16) __bf16 Alds[2][128][LSTR];
  __shared__ __align__(16) __bf16 Blds[2][128][LSTR];

  const int tid  = threadIdx.x;
  const int lane = tid & 31;
  const int wid  = tid >> 5;
  const int wm = wid >> 1, wn = wid & 1;
  const int mblk = blockIdx.y * 128;
  const int nblk = blockIdx.x * 128;
  const int ml = lane & 15;
  const int hh = lane >> 4;

  // cooperative tile staging: thread moves one contiguous 32B slice per tile
  const int lr = tid >> 1;           // row 0..127
  const int lc = (tid & 1) * 16;     // col chunk 0 or 16

  const __bf16* Asrc = A  + (size_t)(mblk + lr) * Kdim + lc;
  const __bf16* Bsrc = Wt + (size_t)(nblk + lr) * Kdim + lc;

  bf16x8 ar0, ar1, br0, br1;
  // preload K-step 0
  ar0 = *reinterpret_cast<const bf16x8*>(Asrc);
  ar1 = *reinterpret_cast<const bf16x8*>(Asrc + 8);
  br0 = *reinterpret_cast<const bf16x8*>(Bsrc);
  br1 = *reinterpret_cast<const bf16x8*>(Bsrc + 8);
  *reinterpret_cast<bf16x8*>(&Alds[0][lr][lc])     = ar0;
  *reinterpret_cast<bf16x8*>(&Alds[0][lr][lc + 8]) = ar1;
  *reinterpret_cast<bf16x8*>(&Blds[0][lr][lc])     = br0;
  *reinterpret_cast<bf16x8*>(&Blds[0][lr][lc + 8]) = br1;
  __syncthreads();

  v8f acc[2][4] = {};
  const int nk = Kdim >> 5;
  int buf = 0;
  for (int kt = 0; kt < nk; ++kt) {
    const bool has_next = (kt + 1) < nk;
    if (has_next) {
      const int k0n = (kt + 1) << 5;
      ar0 = *reinterpret_cast<const bf16x8*>(Asrc + k0n);
      ar1 = *reinterpret_cast<const bf16x8*>(Asrc + k0n + 8);
      br0 = *reinterpret_cast<const bf16x8*>(Bsrc + k0n);
      br1 = *reinterpret_cast<const bf16x8*>(Bsrc + k0n + 8);
    }
    v16bf af[2];
#pragma unroll
    for (int mi = 0; mi < 2; ++mi)
      af[mi] = load_a_frag(&Alds[buf][wm * 32 + mi * 16][0], LSTR, lane, 0);
    v16bf bq[4];
#pragma unroll
    for (int ni = 0; ni < 4; ++ni)
      bq[ni] = load_b_frag(&Blds[buf][wn * 64 + ni * 16][0], LSTR, lane, 0);
#pragma unroll
    for (int mi = 0; mi < 2; ++mi)
#pragma unroll
      for (int ni = 0; ni < 4; ++ni)
        acc[mi][ni] = wmma_bf16(af[mi], bq[ni], acc[mi][ni]);

    if (has_next) {
      const int nb = buf ^ 1;
      *reinterpret_cast<bf16x8*>(&Alds[nb][lr][lc])     = ar0;
      *reinterpret_cast<bf16x8*>(&Alds[nb][lr][lc + 8]) = ar1;
      *reinterpret_cast<bf16x8*>(&Blds[nb][lr][lc])     = br0;
      *reinterpret_cast<bf16x8*>(&Blds[nb][lr][lc + 8]) = br1;
      __syncthreads();
      buf = nb;
    }
  }

  const int m0 = mblk + wm * 32;
  const int n0 = nblk + wn * 64;
#pragma unroll
  for (int mi = 0; mi < 2; ++mi)
#pragma unroll
    for (int ni = 0; ni < 4; ++ni)
#pragma unroll
      for (int r = 0; r < 8; ++r) {
        const int m = m0 + mi * 16 + r + 8 * hh;
        const int n = n0 + ni * 16 + ml;
        const float v = acc[mi][ni][r] + bias[n];
        if (MODE == 0) {
          ((float*)out)[(size_t)m * Ndim + n] = v;
        } else {
          const int b = m >> 11, s = m & (Sc - 1);
          const int h = n >> 6,  d = n & (HDc - 1);
          ((__bf16*)out)[(((size_t)(b * NHc + h)) * Sc + s) * HDc + d] = f2bf(v);
        }
      }
}

// ---------------------------------------------------------------- RoPE (in place)
// one thread per (b,h,s): rotates first 16 dims of q and k in QKV fp32 buffer
__global__ void rope_kernel(float* __restrict__ QKV) {
  const int g = blockIdx.x * blockDim.x + threadIdx.x;   // B*NH*S
  const int s  = g % Sc;
  const int bh = g / Sc;
  const int h  = bh % NHc;
  const int b  = bh / NHc;
  float* q = QKV + ((size_t)(b * Sc + s)) * (3 * Hc) + h * HDc;
  float* k = q + Hc;
  float cs[8], sn[8];
#pragma unroll
  for (int i = 0; i < 8; ++i) {
    const float inv = __powf(ROPEBASE, -(2.0f * i) / (float)RDc);
    const float ang = (float)s * inv;
    cs[i] = __cosf(ang);
    sn[i] = __sinf(ang);
  }
#pragma unroll
  for (int i = 0; i < 8; ++i) {
    const float q1 = q[i], q2 = q[i + 8];
    q[i]     = q1 * cs[i] - q2 * sn[i];
    q[i + 8] = q2 * cs[i] + q1 * sn[i];
    const float k1 = k[i], k2 = k[i + 8];
    k[i]     = k1 * cs[i] - k2 * sn[i];
    k[i + 8] = k2 * cs[i] + k1 * sn[i];
  }
}

// ---------------------------------------------------------------- windowed attention
// one lane per (b,h,s). W=16, HD=64. Outputs A in two bf16 layouts:
//   Ah: (b, s, h*64+d)   row-major  -> input for Wqa/Wka GEMMs
//   AT: (b, h, d, s)                -> V^T fragments for flash P@V
__global__ void window_attn_kernel(const float* __restrict__ QKV,
                                   __bf16* __restrict__ Ah,
                                   __bf16* __restrict__ AT) {
  const int g = blockIdx.x * blockDim.x + threadIdx.x;  // B*NH*S
  const int s  = g % Sc;
  const int bh = g / Sc;
  const int h  = bh % NHc;
  const int b  = bh / NHc;

  const float* qrow = QKV + ((size_t)(b * Sc + s)) * (3 * Hc) + h * HDc;
  float qv[HDc];
#pragma unroll
  for (int t = 0; t < HDc; ++t) qv[t] = qrow[t];

  float sc[Wc];
  float mx = -1e30f;
#pragma unroll
  for (int j = 0; j < Wc; ++j) {
    const int kp = s - (Wc - 1) + j;
    if (kp < 0) { sc[j] = -1e30f; continue; }
    const float* krow = QKV + ((size_t)(b * Sc + kp)) * (3 * Hc) + Hc + h * HDc;
    float d = 0.0f;
    for (int t = 0; t < HDc; ++t) d += qv[t] * krow[t];
    sc[j] = d * SCALEc;
    mx = fmaxf(mx, sc[j]);
  }
  float p[Wc];
  float sum = 0.0f;
#pragma unroll
  for (int j = 0; j < Wc; ++j) {
    p[j] = __expf(sc[j] - mx);   // fully-masked entries underflow to 0
    sum += p[j];
  }
  const float inv = 1.0f / sum;

  for (int t = 0; t < HDc; ++t) {
    float a = 0.0f;
#pragma unroll
    for (int j = 0; j < Wc; ++j) {
      const int kp = s - (Wc - 1) + j;
      if (kp >= 0)
        a += p[j] * QKV[((size_t)(b * Sc + kp)) * (3 * Hc) + 2 * Hc + h * HDc + t];
    }
    a *= inv;
    const __bf16 ab = f2bf(a);
    Ah[((size_t)(b * Sc + s)) * Hc + h * HDc + t] = ab;
    AT[(((size_t)bh) * HDc + t) * Sc + s]         = ab;
  }
}

// ---------------------------------------------------------------- flash causal attention
// qa, ka: bf16 head-major (b,h,s,d); at: bf16 (b,h,d,s); out2: fp32 (b,s,H)
// block = 128 threads = 4 waves; wave owns 16 queries; key tiles of 32.
__global__ __launch_bounds__(128)
void flash_causal_kernel(const __bf16* __restrict__ qa,
                         const __bf16* __restrict__ ka,
                         const __bf16* __restrict__ at,
                         float* __restrict__ out2) {
  __shared__ __align__(16) __bf16 plds[4][16][40];   // per-wave P tile, padded rows

  const int lane = threadIdx.x & 31;
  const int w    = threadIdx.x >> 5;
  const int ml = lane & 15;
  const int hh = lane >> 4;
  const int bh = blockIdx.y;
  const int b  = bh / NHc, h = bh % NHc;
  const int qt = blockIdx.x * 64 + w * 16;

  const __bf16* qbase = qa + ((size_t)bh * Sc + qt) * HDc;
  const __bf16* kbase = ka + (size_t)bh * Sc * HDc;
  const __bf16* vbase = at + (size_t)bh * HDc * Sc;

  const v16bf q0 = load_a_frag(qbase, HDc, lane, 0);
  const v16bf q1 = load_a_frag(qbase, HDc, lane, 32);

  v8f acc[4] = {};
  float mrow[8], lrow[8];
#pragma unroll
  for (int r = 0; r < 8; ++r) { mrow[r] = -1e30f; lrow[r] = 0.0f; }

  for (int j0 = 0; j0 <= qt + 15; j0 += 32) {
    if (j0 + 32 <= qt + 15)
      __builtin_prefetch(kbase + (size_t)(j0 + 32) * HDc, 0, 0);

    float p[2][8];
    // ---- scores: two 16x16 tiles, K-dim = 64 via two WMMAs each
#pragma unroll
    for (int sub = 0; sub < 2; ++sub) {
      const __bf16* kr = kbase + (size_t)(j0 + sub * 16) * HDc;
      const v16bf kf0 = load_b_frag(kr, HDc, lane, 0);
      const v16bf kf1 = load_b_frag(kr, HDc, lane, 32);
      v8f z = {};
      z = wmma_bf16(q0, kf0, z);
      z = wmma_bf16(q1, kf1, z);
      const int key = j0 + sub * 16 + ml;
#pragma unroll
      for (int r = 0; r < 8; ++r) {
        float s = z[r] * SCALEc;
        const int qi = qt + r + 8 * hh;
        if (key > qi) s = -1e30f;          // causal mask
        p[sub][r] = s;
      }
    }
    // ---- online softmax (rows live within a lane-half: xor masks 1/2/4/8)
    float alpha[8];
#pragma unroll
    for (int r = 0; r < 8; ++r) {
      float t = fmaxf(p[0][r], p[1][r]);
      t = fmaxf(t, __shfl_xor(t, 1, 32));
      t = fmaxf(t, __shfl_xor(t, 2, 32));
      t = fmaxf(t, __shfl_xor(t, 4, 32));
      t = fmaxf(t, __shfl_xor(t, 8, 32));
      const float mn = fmaxf(mrow[r], t);
      alpha[r] = __expf(mrow[r] - mn);
      mrow[r]  = mn;
      const float e0 = __expf(p[0][r] - mn);
      const float e1 = __expf(p[1][r] - mn);
      p[0][r] = e0; p[1][r] = e1;
      float s = e0 + e1;
      s += __shfl_xor(s, 1, 32);
      s += __shfl_xor(s, 2, 32);
      s += __shfl_xor(s, 4, 32);
      s += __shfl_xor(s, 8, 32);
      lrow[r] = lrow[r] * alpha[r] + s;
    }
#pragma unroll
    for (int d = 0; d < 4; ++d)
#pragma unroll
      for (int r = 0; r < 8; ++r)
        acc[d][r] *= alpha[r];

    // ---- re-layout P (C-frag order) -> A-frag order through per-wave LDS
#pragma unroll
    for (int sub = 0; sub < 2; ++sub)
#pragma unroll
      for (int r = 0; r < 8; ++r)
        plds[w][r + 8 * hh][sub * 16 + ml] = f2bf(p[sub][r]);
    const v16bf pf = load_a_frag(&plds[w][0][0], 40, lane, 0);

    // ---- P(16x32) @ V(32x64): V^T rows are contiguous in `at`
#pragma unroll
    for (int d = 0; d < 4; ++d) {
      const v16bf vf = load_b_frag(vbase + (size_t)(d * 16) * Sc, Sc, lane, j0);
      acc[d] = wmma_bf16(pf, vf, acc[d]);
    }
  }

#pragma unroll
  for (int d = 0; d < 4; ++d)
#pragma unroll
    for (int r = 0; r < 8; ++r) {
      const int qi = qt + r + 8 * hh;
      const float o = acc[d][r] / lrow[r];
      out2[((size_t)b * Sc + qi) * Hc + h * HDc + d * 16 + ml] = o;
    }
}

// ---------------------------------------------------------------- launch
extern "C" void kernel_launch(void* const* d_in, const int* /*in_sizes*/, int /*n_in*/,
                              void* d_out, int /*out_size*/, void* d_ws, size_t /*ws_size*/,
                              hipStream_t stream) {
  const float* X    = (const float*)d_in[0];
  const float* Wqkv = (const float*)d_in[1];
  const float* bqkv = (const float*)d_in[2];
  const float* Wqa  = (const float*)d_in[3];
  const float* bqa  = (const float*)d_in[4];
  const float* Wka  = (const float*)d_in[5];
  const float* bka  = (const float*)d_in[6];
  const float* Wd   = (const float*)d_in[7];
  const float* bd   = (const float*)d_in[8];

  const int M = Bc * Sc;   // 4096
  char* ws = (char*)d_ws;
  size_t off = 0;
  auto alloc = [&](size_t bytes) -> void* {
    void* p = ws + off;
    off += (bytes + 255) & ~(size_t)255;
    return p;
  };

  __bf16* Xb    = (__bf16*)alloc((size_t)M * Hc * 2);
  __bf16* Wqkvb = (__bf16*)alloc((size_t)3 * Hc * Hc * 2);
  __bf16* Wqab  = (__bf16*)alloc((size_t)Hc * Hc * 2);
  __bf16* Wkab  = (__bf16*)alloc((size_t)Hc * Hc * 2);
  __bf16* Wdb   = (__bf16*)alloc((size_t)Hc * Hc * 2);
  float*  QKV   = (float*) alloc((size_t)M * 3 * Hc * 4);
  __bf16* Ahb   = (__bf16*)alloc((size_t)M * Hc * 2);
  __bf16* ATb   = (__bf16*)alloc((size_t)M * Hc * 2);
  __bf16* QAb   = (__bf16*)alloc((size_t)M * Hc * 2);
  __bf16* KAb   = (__bf16*)alloc((size_t)M * Hc * 2);
  float*  O2    = (float*) alloc((size_t)M * Hc * 4);
  __bf16* O2b   = (__bf16*)alloc((size_t)M * Hc * 2);

  auto cvt = [&](const float* src, __bf16* dst, size_t n) {
    cvt_bf16_kernel<<<dim3((unsigned)((n + 255) / 256)), dim3(256), 0, stream>>>(
        src, dst, (int)n);
  };

  // 1) fp32 -> bf16
  cvt(X,    Xb,    (size_t)M * Hc);
  cvt(Wqkv, Wqkvb, (size_t)3 * Hc * Hc);
  cvt(Wqa,  Wqab,  (size_t)Hc * Hc);
  cvt(Wka,  Wkab,  (size_t)Hc * Hc);
  cvt(Wd,   Wdb,   (size_t)Hc * Hc);

  // 2) QKV = X @ Wqkv^T + bqkv   (fp32 out)
  gemm_bf16_kernel<0><<<dim3(3 * Hc / 128, M / 128), 256, 0, stream>>>(
      Xb, Wqkvb, bqkv, QKV, M, 3 * Hc, Hc);

  // 3) RoPE on q/k (first 16 dims)
  rope_kernel<<<dim3(Bc * NHc * Sc / 256), 256, 0, stream>>>(QKV);

  // 4) sliding-window attention -> A (bf16, two layouts)
  window_attn_kernel<<<dim3(Bc * NHc * Sc / 256), 256, 0, stream>>>(QKV, Ahb, ATb);

  // 5) qa / ka projections (bf16 head-major out)
  gemm_bf16_kernel<1><<<dim3(Hc / 128, M / 128), 256, 0, stream>>>(
      Ahb, Wqab, bqa, QAb, M, Hc, Hc);
  gemm_bf16_kernel<1><<<dim3(Hc / 128, M / 128), 256, 0, stream>>>(
      Ahb, Wkab, bka, KAb, M, Hc, Hc);

  // 6) causal flash attention over A
  flash_causal_kernel<<<dim3(Sc / 64, Bc * NHc), 128, 0, stream>>>(
      QAb, KAb, ATb, O2);

  // 7) final projection -> d_out (fp32)
  cvt(O2, O2b, (size_t)M * Hc);
  gemm_bf16_kernel<0><<<dim3(Hc / 128, M / 128), 256, 0, stream>>>(
      O2b, Wdb, bd, (float*)d_out, M, Hc, Hc);
}
